// TPUGraphNetwork_24927990186156
// MI455X (gfx1250) — matrix operand order repaired and checked by reading, not compiled, not run on hardware
//
#include <hip/hip_runtime.h>

typedef float  v8f   __attribute__((ext_vector_type(8)));
typedef __bf16 v2bf  __attribute__((ext_vector_type(2)));
typedef __bf16 v4bf  __attribute__((ext_vector_type(4)));
typedef __bf16 v8bf  __attribute__((ext_vector_type(8)));
typedef __bf16 v16bf __attribute__((ext_vector_type(16)));

#define CFG 8   // N_CONFIGS

// ---------------- utility ----------------
__global__ void k_zero_i(int* p, int n){ int i=blockIdx.x*blockDim.x+threadIdx.x; if(i<n)p[i]=0; }
__global__ void k_zero_f(float* p, int n){ int i=blockIdx.x*blockDim.x+threadIdx.x; if(i<n)p[i]=0.f; }

// ---------------- degree + CSR ----------------
__global__ void k_deg(const int* __restrict__ src, int* __restrict__ deg, int E){
  int e = blockIdx.x*blockDim.x+threadIdx.x;
  if (e<E) atomicAdd(&deg[src[e]], 1);
}

__global__ void k_block_sums(const int* __restrict__ deg, int* __restrict__ bsum, int N){
  __shared__ int sm[256];
  int i = blockIdx.x*256+threadIdx.x;
  sm[threadIdx.x] = (i<N)?deg[i]:0;
  __syncthreads();
  for (int off=128; off>0; off>>=1){
    if (threadIdx.x<off) sm[threadIdx.x]+=sm[threadIdx.x+off];
    __syncthreads();
  }
  if (threadIdx.x==0) bsum[blockIdx.x]=sm[0];
}

__global__ void k_scan_top(int* bsum, int nb){
  if (blockIdx.x==0 && threadIdx.x==0){
    int run=0;
    for (int i=0;i<nb;i++){ int v=bsum[i]; bsum[i]=run; run+=v; }
  }
}

__global__ void k_scan_apply(const int* __restrict__ deg, const int* __restrict__ bsum,
                             int* __restrict__ row_ptr, int* __restrict__ cursor,
                             float* __restrict__ invdeg, int N, int E){
  __shared__ int sm[256];
  int t = threadIdx.x;
  int i = blockIdx.x*256+t;
  int d = (i<N)?deg[i]:0;
  sm[t]=d; __syncthreads();
  for (int off=1; off<256; off<<=1){
    int add = (t>=off)? sm[t-off] : 0;
    __syncthreads();
    sm[t]+=add;
    __syncthreads();
  }
  int excl = sm[t]-d+bsum[blockIdx.x];
  if (i<N){
    row_ptr[i]=excl; cursor[i]=excl;
    invdeg[i] = 1.0f / fmaxf((float)d, 1.0f);
  }
  if (i==0) row_ptr[N]=E;
}

__global__ void k_fill(const int* __restrict__ src, const int* __restrict__ dst,
                       int* __restrict__ cursor, int* __restrict__ csr_dst, int E){
  int e = blockIdx.x*blockDim.x+threadIdx.x;
  if (e<E){
    int pos = atomicAdd(&cursor[src[e]],1);
    csr_dst[pos]=dst[e];
  }
}

__global__ void k_cum(const int* __restrict__ lengths, int* __restrict__ cum, int G){
  if (blockIdx.x==0 && threadIdx.x==0){
    cum[0]=0;
    for (int g=0; g<G; g++) cum[g+1]=cum[g]+lengths[g];
  }
}

// ---------------- embed + concat -> X0 bf16 [M=N*C, 128] ----------------
__global__ void k_embed(const float* __restrict__ feats, const float* __restrict__ emb,
                        __bf16* __restrict__ X0, int N){
  long r = blockIdx.x;                  // row = n*CFG + c
  int  c = (int)(r & (CFG-1));
  long n = r >> 3;
  const float* f = feats + ((long)c*N + n)*97;
  int d = threadIdx.x;                  // 0..127
  float v;
  if (d < 32){
    int op = (int)f[0];
    op = op < 0 ? 0 : (op > 127 ? 127 : op);
    v = emb[op*32 + d];
  } else {
    v = f[d-31];
  }
  X0[r*128 + d] = (__bf16)v;
}

// -------- swizzle weights [K,64] fp32 -> per-lane WMMA B fragments (bf16) --------
// layout index: ((kb*4+nb)*32 + lane)*16 + i ; element = W[kb*32 + (i>=8?16:0) + (lane>=16?8:0) + i%8][nb*16 + lane%16]
__global__ void k_prep_w(const float* __restrict__ W, __bf16* __restrict__ Bf, int KB){
  int tid = blockIdx.x*blockDim.x+threadIdx.x;
  int total = KB*4*32*16;
  if (tid>=total) return;
  int i    =  tid        & 15;
  int lane = (tid >> 4)  & 31;
  int nb   = (tid >> 9)  & 3;
  int kb   =  tid >> 11;
  int col  = nb*16 + (lane & 15);
  int krow = kb*32 + ((i >> 3) << 4) + ((lane >> 4) << 3) + (i & 7);
  Bf[tid] = (__bf16)W[krow*64 + col];
}

// ---------------- SPMM (mean agg) : one node per block, gather CSR ----------------
__global__ void k_spmm128(const __bf16* __restrict__ Xin, __bf16* __restrict__ Xout,
                          const int* __restrict__ row_ptr, const int* __restrict__ csr_dst,
                          const float* __restrict__ invdeg){
  int n = blockIdx.x, t = threadIdx.x;          // 256 threads, C*D = 1024 elems -> 4/thread
  float a0=0.f,a1=0.f,a2=0.f,a3=0.f;
  int s = row_ptr[n], e = row_ptr[n+1];
  for (int j=s;j<e;j++){
    const v4bf v = *(const v4bf*)(Xin + (long)csr_dst[j]*1024 + t*4);
    a0 += (float)v[0]; a1 += (float)v[1]; a2 += (float)v[2]; a3 += (float)v[3];
  }
  float sc = invdeg[n];
  v4bf o;
  o[0]=(__bf16)(a0*sc); o[1]=(__bf16)(a1*sc); o[2]=(__bf16)(a2*sc); o[3]=(__bf16)(a3*sc);
  *(v4bf*)(Xout + (long)n*1024 + t*4) = o;
}

__global__ void k_spmm64(const __bf16* __restrict__ Xin, __bf16* __restrict__ Xout,
                         const int* __restrict__ row_ptr, const int* __restrict__ csr_dst,
                         const float* __restrict__ invdeg){
  int n = blockIdx.x, t = threadIdx.x;          // C*D = 512 elems -> 2/thread
  float a0=0.f,a1=0.f;
  int s = row_ptr[n], e = row_ptr[n+1];
  for (int j=s;j<e;j++){
    const v2bf v = *(const v2bf*)(Xin + (long)csr_dst[j]*512 + t*2);
    a0 += (float)v[0]; a1 += (float)v[1];
  }
  float sc = invdeg[n];
  v2bf o;
  o[0]=(__bf16)(a0*sc); o[1]=(__bf16)(a1*sc);
  *(v2bf*)(Xout + (long)n*512 + t*2) = o;
}

// ---------------- GEMM [M,K] x [K,64] + bias + relu, bf16 in, bf16 out ----------------
// one wave per 16-row M-tile; 4 accumulators cover N=64; K = KB*32 via v_wmma_f32_16x16x32_bf16
template<int KB>
__global__ __launch_bounds__(256)
void k_gemm(const __bf16* __restrict__ A, const __bf16* __restrict__ Bf,
            const float* __restrict__ bias, __bf16* __restrict__ Out, int Mtiles){
  const int K = KB*32;
  int wave = threadIdx.x >> 5;
  int lane = threadIdx.x & 31;
  int mtile = blockIdx.x*(blockDim.x >> 5) + wave;
  if (mtile >= Mtiles) return;                 // wave-uniform: EXEC stays all-ones for WMMA
  long row = (long)mtile*16 + (lane & 15);
  v8f acc[4] = {};
  for (int kb=0; kb<KB; kb++){
    // A fragment (16-bit A 16x32 layout): lane<16 -> K {0..7,16..23}; lane>=16 -> K {8..15,24..31}
    int off1 = kb*32 + ((lane >> 4) << 3);
    const __bf16* ap = A + row*K + off1;
    v8bf q1 = *(const v8bf*)(ap);
    v8bf q2 = *(const v8bf*)(ap + 16);
    v16bf a;
#pragma unroll
    for (int i=0;i<8;i++){ a[i]=q1[i]; a[i+8]=q2[i]; }
#pragma unroll
    for (int nb=0; nb<4; nb++){
      v16bf b = *(const v16bf*)(Bf + ((kb*4+nb)*32 + lane)*16);
      acc[nb] = __builtin_amdgcn_wmma_f32_16x16x32_bf16(
          false, a, false, b, (short)0, acc[nb], false, false);
    }
  }
  // C/D layout: VGPR i -> M = i + 8*(lane>=16), N = lane&15
  long mbase = (long)mtile*16 + ((lane >> 4) << 3);
  int ncol = lane & 15;
#pragma unroll
  for (int nb=0; nb<4; nb++){
    float bc = bias[nb*16 + ncol];
#pragma unroll
    for (int i=0;i<8;i++){
      float v = acc[nb][i] + bc;
      v = fmaxf(v, 0.0f);
      Out[(mbase+i)*64 + nb*16 + ncol] = (__bf16)v;
    }
  }
}

// ---------------- projection |h@Wp + bp| + per-graph pooling ----------------
__global__ void k_proj_pool(const __bf16* __restrict__ H, const float* __restrict__ Wp,
                            const float* __restrict__ bp, const int* __restrict__ cum,
                            float* __restrict__ out, long M, int G){
  __shared__ float sW[64];
  __shared__ float pool[128];
  int t = threadIdx.x;
  if (t < 64) sW[t] = Wp[t];
  if (t < 128) pool[t] = 0.f;
  __syncthreads();
  long r = (long)blockIdx.x*blockDim.x + t;
  if (r < M){
    const __bf16* h = H + r*64;
    float s = 0.f;
#pragma unroll
    for (int dv=0; dv<8; dv++){
      v8bf hv = *(const v8bf*)(h + dv*8);
#pragma unroll
      for (int i=0;i<8;i++) s += (float)hv[i]*sW[dv*8+i];
    }
    s = fabsf(s + bp[0]);
    long n = r >> 3; int c = (int)(r & 7);
    int g = G-1;
    for (int gg=0; gg<G-1; gg++){ if (n < (long)cum[gg+1]) { g = gg; break; } }
    atomicAdd(&pool[g*CFG + c], s);
  }
  __syncthreads();
  if (t < G*CFG){
    float v = pool[t];
    if (v != 0.f) atomicAdd(&out[t], v);
  }
}

// ---------------- host ----------------
extern "C" void kernel_launch(void* const* d_in, const int* in_sizes, int n_in,
                              void* d_out, int out_size, void* d_ws, size_t ws_size,
                              hipStream_t stream) {
  (void)n_in; (void)ws_size;
  const float* feats  = (const float*)d_in[0];
  const int*   edges  = (const int*)  d_in[1];   // [2,E]: row0=src, row1=dst
  const int*   lens   = (const int*)  d_in[2];
  const float* emb    = (const float*)d_in[3];
  const float* W1     = (const float*)d_in[4];
  const float* b1     = (const float*)d_in[5];
  const float* W2     = (const float*)d_in[6];
  const float* b2     = (const float*)d_in[7];
  const float* Wp     = (const float*)d_in[8];
  const float* bp     = (const float*)d_in[9];
  float* out = (float*)d_out;

  const int  N = in_sizes[0] / (CFG*97);
  const int  E = in_sizes[1] / 2;
  const int  G = in_sizes[2];
  const long M = (long)N * CFG;
  const int  Mtiles = (int)(M / 16);

  // workspace carve-up (256B aligned)
  char* base = (char*)d_ws; size_t off = 0;
  auto take = [&](size_t bytes)->void* {
    void* p = base + off; off = (off + bytes + 255) & ~(size_t)255; return p;
  };
  __bf16* X0   = (__bf16*)take((size_t)M*128*2);  // x0, later reused as agg2
  __bf16* BUF2 = (__bf16*)take((size_t)M*128*2);  // agg1, later reused as h2
  __bf16* H1   = (__bf16*)take((size_t)M*64*2);
  int*   deg     = (int*)  take((size_t)N*4);
  int*   row_ptr = (int*)  take((size_t)(N+1)*4);
  int*   cursor  = (int*)  take((size_t)N*4);
  float* invdeg  = (float*)take((size_t)N*4);
  int nbk = (N+255)/256;
  int*   bsum    = (int*)  take((size_t)nbk*4);
  int*   csr_dst = (int*)  take((size_t)E*4);
  int*   cum     = (int*)  take((size_t)(G+1)*4);
  __bf16* Wf1 = (__bf16*)take((size_t)4*4*32*16*2);
  __bf16* Wf2 = (__bf16*)take((size_t)2*4*32*16*2);

  __bf16* AGG1 = BUF2;
  __bf16* AGG2 = X0;
  __bf16* H2   = BUF2;

  // graph prep
  k_zero_i<<<(N+255)/256, 256, 0, stream>>>(deg, N);
  k_zero_f<<<1, 256, 0, stream>>>(out, out_size);
  k_deg<<<(E+255)/256, 256, 0, stream>>>(edges, deg, E);
  k_block_sums<<<nbk, 256, 0, stream>>>(deg, bsum, N);
  k_scan_top<<<1, 1, 0, stream>>>(bsum, nbk);
  k_scan_apply<<<nbk, 256, 0, stream>>>(deg, bsum, row_ptr, cursor, invdeg, N, E);
  k_fill<<<(E+255)/256, 256, 0, stream>>>(edges, edges+E, cursor, csr_dst, E);
  k_cum<<<1, 1, 0, stream>>>(lens, cum, G);

  // weights -> WMMA fragment layout
  k_prep_w<<<(8192+255)/256, 256, 0, stream>>>(W1, Wf1, 4);
  k_prep_w<<<(4096+255)/256, 256, 0, stream>>>(W2, Wf2, 2);

  // embed
  k_embed<<<(unsigned)M, 128, 0, stream>>>(feats, emb, X0, N);

  // layer 1
  k_spmm128<<<N, 256, 0, stream>>>(X0, AGG1, row_ptr, csr_dst, invdeg);
  k_gemm<4><<<(Mtiles+7)/8, 256, 0, stream>>>(AGG1, Wf1, b1, H1, Mtiles);

  // layer 2
  k_spmm64<<<N, 256, 0, stream>>>(H1, AGG2, row_ptr, csr_dst, invdeg);
  k_gemm<2><<<(Mtiles+7)/8, 256, 0, stream>>>(AGG2, Wf2, b2, H2, Mtiles);

  // projection + pooling
  k_proj_pool<<<(unsigned)((M+255)/256), 256, 0, stream>>>(H2, Wp, bp, cum, out, M, G);
}
